// Val2Cst_Layer_38190849196759
// MI455X (gfx1250) — matrix operand order repaired and checked by reading, not compiled, not run on hardware
//
#include <hip/hip_runtime.h>

#define HID   128
#define NVAL  100000
#define NCST  20000
#define NEDGE 500000

typedef __attribute__((ext_vector_type(16))) __bf16 v16bf;
typedef __attribute__((ext_vector_type(8)))  __bf16 v8bf;
typedef __attribute__((ext_vector_type(8)))  float  v8f;

// A operand (16x32 bf16 tile of chunk kc) from LDS, row-major stride HID.
// Layout: row = lane&15, K-half = lane>>4; elems 0..15 = K(kc*32 + half*16 + 0..15).
__device__ __forceinline__ v16bf load_a_lds(const __bf16* p, int row, int kc, int half) {
  const __bf16* base = p + row * HID + kc * 32 + half * 16;
  v8bf lo = *(const v8bf*)(base);
  v8bf hi = *(const v8bf*)(base + 8);
  return __builtin_shufflevector(lo, hi, 0,1,2,3,4,5,6,7,8,9,10,11,12,13,14,15);
}

// B operand (32x16 bf16 tile of chunk kc) from a transposed [N][K] bf16 weight.
// Layout: col = lane&15, K-half = lane>>4; 16 contiguous K values (32B -> 2x b128 loads).
__device__ __forceinline__ v16bf load_b_glb(const __bf16* __restrict__ Wt, int ncol, int kc, int half) {
  return *(const v16bf*)(Wt + (size_t)ncol * HID + kc * 32 + half * 16);
}

__global__ void prep_weights_kernel(const float* __restrict__ W1, const float* __restrict__ W2,
                                    const float* __restrict__ W3,
                                    __bf16* __restrict__ W1t, __bf16* __restrict__ W2t,
                                    __bf16* __restrict__ W3t) {
  int i = blockIdx.x * blockDim.x + threadIdx.x;
  if (i < HID * HID) {                       // W1t[n][k] = W1[k][n] (first 128 rows only)
    int n = i / HID, k = i % HID;
    W1t[i] = (__bf16)W1[k * HID + n];
  } else if (i < 2 * HID * HID) {            // W2t[n][k] = W2[k][n]
    int j = i - HID * HID;
    int n = j / HID, k = j % HID;
    W2t[j] = (__bf16)W2[k * HID + n];
  } else if (i < 6 * HID * HID) {            // W3t[n][k] = W3[k][n], n < 512
    int j = i - 2 * HID * HID;
    int n = j / HID, k = j % HID;
    W3t[j] = (__bf16)W3[k * (4 * HID) + n];
  }
}

__global__ void zero_kernel(float* __restrict__ p, int n) {
  int i = blockIdx.x * blockDim.x + threadIdx.x;
  if (i < n) p[i] = 0.f;
}

__global__ __launch_bounds__(32) void fused_val_kernel(
    const float* __restrict__ h_val, const float* __restrict__ assign,
    const float* __restrict__ W1, const float* __restrict__ b1,
    const __bf16* __restrict__ W1t, const __bf16* __restrict__ W2t, const __bf16* __restrict__ W3t,
    const float* __restrict__ ln1g, const float* __restrict__ ln1b,
    const float* __restrict__ ln2g, const float* __restrict__ ln2b,
    float* __restrict__ xval_out, __bf16* __restrict__ m_out)
{
  __shared__ __align__(32) __bf16 lds_a[16 * HID];       // x tile (bf16)
  __shared__ __align__(32) __bf16 lds_t[16 * HID];       // relu(t) / x_val (bf16)
  __shared__ __align__(16) float  lds_f[16 * 4 * HID];   // fp32 GEMM outputs (16x512)
  __shared__ float lds_s[16];
  __shared__ float lds_mu[16];
  __shared__ float lds_rs[16];

  const int lane = threadIdx.x;
  const int l16  = lane & 15;
  const int half = lane >> 4;
  const int row0 = blockIdx.x * 16;   // NVAL = 16 * 6250 exactly

  // ---- stage x tile -> bf16 LDS, assign -> LDS ----
  {
    const float4* src = (const float4*)(h_val + (size_t)row0 * HID);
    for (int i = lane; i < 16 * HID / 4; i += 32) {
      float4 v = src[i];
      int o = i * 4;
      lds_a[o + 0] = (__bf16)v.x; lds_a[o + 1] = (__bf16)v.y;
      lds_a[o + 2] = (__bf16)v.z; lds_a[o + 3] = (__bf16)v.w;
    }
    if (lane < 16) lds_s[lane] = assign[row0 + lane];
  }
  __syncthreads();

  // ---- GEMM1: t = relu(x @ W1[:128] + assign * W1[128] + b1) -> lds_t (bf16) ----
  for (int nt = 0; nt < 8; ++nt) {
    v8f acc = {};
#pragma unroll
    for (int kc = 0; kc < 4; ++kc)
      acc = __builtin_amdgcn_wmma_f32_16x16x32_bf16(
          false, load_a_lds(lds_a, l16, kc, half),
          false, load_b_glb(W1t, nt * 16 + l16, kc, half),
          (short)0, acc, false, false);
    const float bv = b1[nt * 16 + l16];
    const float wl = W1[HID * HID + nt * 16 + l16];  // last row of W1 (assign column)
#pragma unroll
    for (int v = 0; v < 8; ++v) {
      const int m = v + 8 * half;                    // C/D layout: row = vgpr + 8*half
      float r = acc[v] + bv + lds_s[m] * wl;
      lds_t[m * HID + nt * 16 + l16] = (__bf16)fmaxf(r, 0.f);
    }
  }
  __syncthreads();

  // ---- GEMM2: y = t @ W2 -> lds_f (fp32) ----
  for (int nt = 0; nt < 8; ++nt) {
    v8f acc = {};
#pragma unroll
    for (int kc = 0; kc < 4; ++kc)
      acc = __builtin_amdgcn_wmma_f32_16x16x32_bf16(
          false, load_a_lds(lds_t, l16, kc, half),
          false, load_b_glb(W2t, nt * 16 + l16, kc, half),
          (short)0, acc, false, false);
#pragma unroll
    for (int v = 0; v < 8; ++v)
      lds_f[(v + 8 * half) * HID + nt * 16 + l16] = acc[v];
  }
  __syncthreads();

  // ---- LN1 over 128 ----
  if (lane < 16) {
    float s = 0.f, ss = 0.f;
    for (int c = 0; c < HID; ++c) { float x = lds_f[lane * HID + c]; s += x; ss += x * x; }
    float mu  = s * (1.0f / HID);
    float var = ss * (1.0f / HID) - mu * mu;
    lds_mu[lane] = mu;
    lds_rs[lane] = rsqrtf(var + 1e-5f);
  }
  __syncthreads();
  for (int i = lane; i < 16 * HID; i += 32) {
    int r = i >> 7, c = i & (HID - 1);
    float x = (lds_f[i] - lds_mu[r]) * lds_rs[r] * ln1g[c] + ln1b[c];
    xval_out[(size_t)(row0 + r) * HID + c] = x;   // x_val output (fp32)
    lds_t[i] = (__bf16)x;                         // A operand for GEMM3
  }
  __syncthreads();

  // ---- GEMM3: z = x_val @ W3 -> lds_f (16 x 512 fp32) ----
  for (int nt = 0; nt < 32; ++nt) {
    v8f acc = {};
#pragma unroll
    for (int kc = 0; kc < 4; ++kc)
      acc = __builtin_amdgcn_wmma_f32_16x16x32_bf16(
          false, load_a_lds(lds_t, l16, kc, half),
          false, load_b_glb(W3t, nt * 16 + l16, kc, half),
          (short)0, acc, false, false);
#pragma unroll
    for (int v = 0; v < 8; ++v)
      lds_f[(v + 8 * half) * (4 * HID) + nt * 16 + l16] = acc[v];
  }
  __syncthreads();

  // ---- LN2 over 512, write m_val as bf16 ([V][512] == [4V][128] flat; 102 MB, L2-resident) ----
  if (lane < 16) {
    float s = 0.f, ss = 0.f;
    for (int c = 0; c < 4 * HID; ++c) { float x = lds_f[lane * 4 * HID + c]; s += x; ss += x * x; }
    float mu  = s * (1.0f / (4 * HID));
    float var = ss * (1.0f / (4 * HID)) - mu * mu;
    lds_mu[lane] = mu;
    lds_rs[lane] = rsqrtf(var + 1e-5f);
  }
  __syncthreads();
  for (int i = lane; i < 16 * 4 * HID; i += 32) {
    int r = i >> 9, c = i & (4 * HID - 1);
    float x = (lds_f[i] - lds_mu[r]) * lds_rs[r] * ln2g[c] + ln2b[c];
    m_out[(size_t)(row0 + r) * (4 * HID) + c] = (__bf16)x;
  }
}

// 16 lanes per edge: each lane gathers 16B (v8bf) of the message row (coalesced per
// half-wave, mostly L2 hits since m is 102 MB < 192 MB L2), converts to fp32 and
// scatter-sums with global_atomic_add_f32 into the constraint row.
__global__ void scatter_kernel(const int* __restrict__ cst_src, const int* __restrict__ cst_dst,
                               const int* __restrict__ LE, const int* __restrict__ PE,
                               const __bf16* __restrict__ m, float* __restrict__ out) {
  long long t = (long long)blockIdx.x * blockDim.x + threadIdx.x;
  int e = (int)(t >> 4);
  int q = (int)(t & 15);
  if (e < NEDGE) {
    int row = 4 * cst_dst[e] + 2 * LE[e] + PE[e];
    v8bf v = *(const v8bf*)(m + (size_t)row * HID + q * 8);
    float* dst = out + (size_t)cst_src[e] * HID + q * 8;
#pragma unroll
    for (int j = 0; j < 8; ++j)
      atomicAdd(dst + j, (float)v[j]);
  }
}

extern "C" void kernel_launch(void* const* d_in, const int* in_sizes, int n_in,
                              void* d_out, int out_size, void* d_ws, size_t ws_size,
                              hipStream_t stream) {
  const float* h_val  = (const float*)d_in[0];
  const float* assign = (const float*)d_in[1];
  const int*   cst_src = (const int*)d_in[2];
  const int*   cst_dst = (const int*)d_in[3];
  const int*   LE      = (const int*)d_in[4];
  const int*   PE      = (const int*)d_in[5];
  const float* W1   = (const float*)d_in[6];
  const float* b1   = (const float*)d_in[7];
  const float* W2   = (const float*)d_in[8];
  const float* ln1g = (const float*)d_in[9];
  const float* ln1b = (const float*)d_in[10];
  const float* W3   = (const float*)d_in[11];
  const float* ln2g = (const float*)d_in[12];
  const float* ln2b = (const float*)d_in[13];
  (void)in_sizes; (void)n_in; (void)out_size; (void)ws_size;

  float* out = (float*)d_out;                 // [NCST*128] r_cst, then [NVAL*128] x_val
  char*  ws  = (char*)d_ws;
  __bf16* W1t = (__bf16*)(ws);                          // 32 KB
  __bf16* W2t = (__bf16*)(ws + 32768);                  // 32 KB
  __bf16* W3t = (__bf16*)(ws + 65536);                  // 128 KB
  __bf16* m_val = (__bf16*)(ws + 65536 + 131072);       // 4*NVAL*128 bf16 (~102 MB)

  // 1. bf16 transposed weights
  prep_weights_kernel<<<(6 * HID * HID + 255) / 256, 256, 0, stream>>>(W1, W2, W3, W1t, W2t, W3t);

  // 2. zero r_cst accumulator region (atomics accumulate into it)
  zero_kernel<<<(NCST * HID + 255) / 256, 256, 0, stream>>>(out, NCST * HID);

  // 3. fused encoder: one wave per 16-row tile
  fused_val_kernel<<<NVAL / 16, 32, 0, stream>>>(
      h_val, assign, W1, b1, W1t, W2t, W3t, ln1g, ln1b, ln2g, ln2b,
      out + (size_t)NCST * HID, m_val);

  // 4. edge gather + atomic scatter-sum (16 lanes / edge)
  long long nthr = (long long)NEDGE * 16;
  scatter_kernel<<<(unsigned)((nthr + 255) / 256), 256, 0, stream>>>(
      cst_src, cst_dst, LE, PE, m_val, out);
}
